// CycleFC_2954937500320
// MI455X (gfx1250) — matrix operand order repaired and checked by reading, not compiled, not run on hardware
//
#include <hip/hip_runtime.h>

// ---------------------------------------------------------------------------
// CycleFC on MI455X (gfx1250):
//   out[b,o,h,w] = sum_c weight[o,c] * x[b,c,h,w+s(c)] + bias[o]
//   s(c) = ((c+3)%7)-3, zero padded at W edges.
// GEMM view: M=256 (C_out), N=64*56*56=200704 (b,h,w), K=256 (C_in).
// Roofline: 411 MB HBM traffic -> ~17.6us floor @ 23.3 TB/s; bf16 WMMA
// (f32 accum) compute ~5us -> memory bound, so bf16 is the right precision.
// Structure: 128x128x32 tiles, double-buffered LDS (1 barrier/K-step, global
// prefetch hidden under WMMA), incremental mod-7 shift (no per-element div),
// packed cvt_pk_bf16 + 2addr DS stores, NT stores for the write-once output
// so the 192MB L2 stays reserved for the x tile shared by sibling M-blocks.
// ---------------------------------------------------------------------------

typedef __bf16 bf16;
typedef bf16  v16bf __attribute__((ext_vector_type(16)));
typedef bf16  v8bf  __attribute__((ext_vector_type(8)));
typedef bf16  v2bf  __attribute__((ext_vector_type(2)));
typedef float v8f   __attribute__((ext_vector_type(8)));

#define B_    64
#define C_    256
#define H_    56
#define W_    56
#define HW_   (H_ * W_)        // 3136
#define NTOT  (B_ * HW_)       // 200704
#define BM    128
#define BN    128
#define BK    32
#define LDT   40               // LDS row stride in bf16 (80B, 16B-aligned)

__global__ __launch_bounds__(256)
void cyclefc_wmma_kernel(const float* __restrict__ x,
                         const float* __restrict__ wgt,
                         const float* __restrict__ bias,
                         float* __restrict__ out)
{
    __shared__ bf16 Alds[2][BM * LDT];   // weight tiles, [m][k]
    __shared__ bf16 Blds[2][BN * LDT];   // shifted-x tiles, [n][k]

    const int t    = threadIdx.x;
    const int lane = t & 31;
    const int wid  = t >> 5;          // 0..7
    const int wm   = wid >> 2;        // 0..1  (64-row band)
    const int wn   = wid & 3;         // 0..3  (32-col band)
    const int half = lane >> 4;
    const int l16  = lane & 15;

    const int m0 = blockIdx.x * BM;   // x fastest -> sibling M-tiles share x in L2
    const int n0 = blockIdx.y * BN;

    // ---- B staging: thread owns one n; channel pairs c = kb + 2*pb + 4*j + {0,1}
    const int bn_local = t & 127;
    const int pb       = t >> 7;                 // 0 or 1
    const int n_g   = n0 + bn_local;
    const int b_img = n_g / HW_;
    const int hw_g  = n_g - b_img * HW_;
    const int h_g   = hw_g / W_;
    const int w_g   = hw_g - h_g * W_;
    const int xrow0 = b_img * (C_ * HW_) + h_g * W_;   // (b, c=0, h, w=0)

    // ---- A staging: thread owns contiguous c pair; rows m = (t>>4) + 16*j
    const int acp = (t & 15) * 2;
    const int am0 = t >> 4;

    float  bP[16];      // prefetched shifted-x values
    float2 aP[8];       // prefetched weight pairs

    // Global->regs for K-step kb. Shift s(c) = r-3 with r=(c+3)%7 maintained
    // incrementally: one uniform kb%7, then +4 mod 7 per unrolled step.
    auto load_tiles = [&](int kb) {
        #pragma unroll
        for (int j = 0; j < 8; ++j) {
            const int ml = am0 + 16 * j;
            aP[j] = *(const float2*)&wgt[(m0 + ml) * C_ + kb + acp];
        }
        int r = (kb % 7) + 2 * pb + 3;           // residue of (c0 + 3) mod 7
        if (r >= 7) r -= 7;
        int base = xrow0 + (kb + 2 * pb) * HW_ + w_g - 3;
        #pragma unroll
        for (int j = 0; j < 8; ++j) {
            const int r1 = (r == 6) ? 0 : r + 1;
            float v0 = 0.0f, v1 = 0.0f;
            if ((unsigned)(w_g + r  - 3) < (unsigned)W_) v0 = x[base + r];
            if ((unsigned)(w_g + r1 - 3) < (unsigned)W_) v1 = x[base + HW_ + r1];
            bP[2 * j]     = v0;
            bP[2 * j + 1] = v1;
            base += 4 * HW_;                     // folds into load ioffsets
            r += 4; if (r >= 7) r -= 7;
        }
    };

    // Regs -> LDS (bf16, packed pair stores -> v_cvt_pk_bf16_f32 + ds 2addr)
    auto store_tiles = [&](int buf) {
        bf16* Al = Alds[buf];
        bf16* Bl = Blds[buf];
        #pragma unroll
        for (int j = 0; j < 8; ++j) {
            v2bf pk = { (bf16)aP[j].x, (bf16)aP[j].y };
            *(v2bf*)&Al[(am0 + 16 * j) * LDT + acp] = pk;
        }
        #pragma unroll
        for (int j = 0; j < 8; ++j) {
            v2bf pk = { (bf16)bP[2 * j], (bf16)bP[2 * j + 1] };
            *(v2bf*)&Bl[bn_local * LDT + 2 * pb + 4 * j] = pk;
        }
    };

    v8f acc[4][2];
    #pragma unroll
    for (int mi = 0; mi < 4; ++mi)
        #pragma unroll
        for (int ni = 0; ni < 2; ++ni)
            acc[mi][ni] = v8f{};

    load_tiles(0);
    int buf = 0;
    for (int kb = 0; kb < C_; kb += BK) {
        store_tiles(buf);
        __syncthreads();                 // s_wait_dscnt 0 + barrier: ping-pong safe
        if (kb + BK < C_) load_tiles(kb + BK);   // latency hidden under WMMAs

        const bf16* Al = Alds[buf];
        const bf16* Bl = Blds[buf];
        v16bf afr[4], bfr[2];
        #pragma unroll
        for (int mi = 0; mi < 4; ++mi) {
            const int row = wm * 64 + mi * 16 + l16;
            const int k0  = half ? 8 : 0;        // lanes16-31: K 8-15 / 24-31
            v8bf lo = *(const v8bf*)&Al[row * LDT + k0];
            v8bf hi = *(const v8bf*)&Al[row * LDT + k0 + 16];
            afr[mi] = __builtin_shufflevector(lo, hi,
                0,1,2,3,4,5,6,7,8,9,10,11,12,13,14,15);
        }
        #pragma unroll
        for (int ni = 0; ni < 2; ++ni) {
            const int col = wn * 32 + ni * 16 + l16;
            const int k0  = half ? 8 : 0;        // lanes16-31: K 8-15 / 24-31
            v8bf lo = *(const v8bf*)&Bl[col * LDT + k0];
            v8bf hi = *(const v8bf*)&Bl[col * LDT + k0 + 16];
            bfr[ni] = __builtin_shufflevector(lo, hi,
                0,1,2,3,4,5,6,7,8,9,10,11,12,13,14,15);
        }
        #pragma unroll
        for (int mi = 0; mi < 4; ++mi)
            #pragma unroll
            for (int ni = 0; ni < 2; ++ni)
                acc[mi][ni] = __builtin_amdgcn_wmma_f32_16x16x32_bf16(
                    false, afr[mi], false, bfr[ni],
                    (short)0, acc[mi][ni], false, false);
        buf ^= 1;
    }

    // ---- epilogue: C/D layout (VGPR r <-> M = r + 8*half, lane <-> N) -> NCHW
    // NT stores: out is write-once/never-read; don't displace x lines in L2.
    #pragma unroll
    for (int ni = 0; ni < 2; ++ni) {
        const int n_e  = n0 + wn * 32 + ni * 16 + l16;
        const int b_e  = n_e / HW_;
        const int hw_e = n_e - b_e * HW_;
        const int obase_n = b_e * (C_ * HW_) + hw_e;
        #pragma unroll
        for (int mi = 0; mi < 4; ++mi) {
            const int obase = m0 + wm * 64 + mi * 16 + half * 8;  // mult of 8
            float bb[8];
            *(float4*)&bb[0] = *(const float4*)&bias[obase];
            *(float4*)&bb[4] = *(const float4*)&bias[obase + 4];
            #pragma unroll
            for (int r = 0; r < 8; ++r)
                __builtin_nontemporal_store(acc[mi][ni][r] + bb[r],
                                            &out[obase_n + (obase + r) * HW_]);
        }
    }
}

extern "C" void kernel_launch(void* const* d_in, const int* in_sizes, int n_in,
                              void* d_out, int out_size, void* d_ws, size_t ws_size,
                              hipStream_t stream) {
    (void)in_sizes; (void)n_in; (void)out_size; (void)d_ws; (void)ws_size;
    const float* x    = (const float*)d_in[0];
    const float* wgt  = (const float*)d_in[1];
    const float* bias = (const float*)d_in[2];
    float* out        = (float*)d_out;

    dim3 grid(C_ / BM, NTOT / BN);   // (2, 1568): M tiles adjacent for L2 reuse
    cyclefc_wmma_kernel<<<grid, 256, 0, stream>>>(x, wgt, bias, out);
}
